// DS2_32813550141552
// MI455X (gfx1250) — matrix operand order repaired and checked by reading, not compiled, not run on hardware
//
#include <hip/hip_runtime.h>
#include <hip/hip_bf16.h>
#include <math.h>

typedef __attribute__((ext_vector_type(16))) __bf16 v16bf;
typedef __attribute__((ext_vector_type(8)))  __bf16 v8bf;
typedef __attribute__((ext_vector_type(4)))  __bf16 v4bf;
typedef __attribute__((ext_vector_type(8)))  float  v8f;
typedef __attribute__((ext_vector_type(4)))  float  v4f;
typedef __attribute__((ext_vector_type(4)))  unsigned int v4u;

#define BATCH 32
#define TIN   2000
#define FIN   80
#define TOUT  995
#define CCH   512
#define UU    512
#define N3    1536
#define NV    29
#define ROWS  (BATCH*TOUT)     /* 31840 */
#define KT16  16               /* K tiles of 32 over K=512 */
#define NT96  96               /* N tiles of 16 over N=1536 */
#define BN_EPS 0.001f

__device__ inline v16bf join8(v8bf lo, v8bf hi) {
  v16bf r;
#pragma unroll
  for (int i = 0; i < 8; ++i) { r[i] = lo[i]; r[i + 8] = hi[i]; }
  return r;
}

__device__ inline v8f wmma_bf16(v16bf a, v16bf b, v8f c) {
  return __builtin_amdgcn_wmma_f32_16x16x32_bf16(false, a, false, b, (short)0, c,
                                                 false, false);
}

__device__ inline float sigmoidf(float x) { return 1.f / (1.f + __expf(-x)); }

// ---------------------------------------------------------------------------
// Pack a 512x1536 f32 row-major matrix into bf16 WMMA B-fragment-major layout:
// blocks [kt(16)][nt(96)] of 1KB; within a block, lane-major 32B:
//   lane n (0..15):  Vj = { B[kt*32+2j+(j>=4?8:0)][nt*16+n],  B[kt*32+2j+(j>=4?8:0)+1][..] }
//   lane n+16:       Vj = { B[kt*32+8+2j+(j>=4?8:0)][..],  B[kt*32+9+2j+(j>=4?8:0)][..] }
// ---------------------------------------------------------------------------
__global__ void pack_weights(const float* __restrict__ kf, const float* __restrict__ kb,
                             const float* __restrict__ rkf, const float* __restrict__ rkb,
                             __bf16* __restrict__ pkf, __bf16* __restrict__ pkb,
                             __bf16* __restrict__ prkf, __bf16* __restrict__ prkb) {
  const long per = (long)KT16 * NT96 * 32 * 8;  // 393216 dword-pairs per matrix
  long id = (long)blockIdx.x * 256 + threadIdx.x;
  if (id >= 4 * per) return;
  int mat = (int)(id / per);
  long r  = id % per;
  const float* src = (mat == 0) ? kf : (mat == 1) ? kb : (mat == 2) ? rkf : rkb;
  __bf16*      dst = (mat == 0) ? pkf : (mat == 1) ? pkb : (mat == 2) ? prkf : prkb;
  int jj = (int)(r & 7);
  int ln = (int)((r >> 3) & 31);
  int blk = (int)(r >> 8);
  int nt = blk % NT96, kt = blk / NT96;
  int n = nt * 16 + (ln & 15);
  int k = kt * 32 + (ln >> 4) * 8 + jj * 2 + (jj >= 4 ? 8 : 0);
  dst[(long)blk * 512 + ln * 16 + jj * 2]     = (__bf16)src[(long)k * N3 + n];
  dst[(long)blk * 512 + ln * 16 + jj * 2 + 1] = (__bf16)src[(long)(k + 1) * N3 + n];
}

// ---------------------------------------------------------------------------
// Conv1D (K=11, stride 2, VALID) + bias + ReLU + BN.  grid (125, 32), block 512.
// ---------------------------------------------------------------------------
__global__ __launch_bounds__(512) void conv_bn(
    const float* __restrict__ x, const float* __restrict__ wconv,
    const float* __restrict__ cb, const float* __restrict__ gamma,
    const float* __restrict__ beta, const float* __restrict__ mean,
    const float* __restrict__ var, float* __restrict__ outf,
    __bf16* __restrict__ outb) {
  __shared__ float xs[25 * 80];  // window rows 2*t0 .. 2*t0+24
  const int b = blockIdx.y, t0 = blockIdx.x * 8, c = threadIdx.x;
  for (int i = c; i < 25 * 80; i += 512) {
    int rr = 2 * t0 + i / 80, ff = i % 80;
    xs[i] = (rr < TIN) ? x[((long)b * TIN + rr) * FIN + ff] : 0.f;
  }
  __syncthreads();
  float acc[8];
  float cbv = cb[c];
#pragma unroll
  for (int i = 0; i < 8; ++i) acc[i] = cbv;
  for (int k = 0; k < 11; ++k)
    for (int f = 0; f < 80; f += 4) {
      const float* wp = wconv + ((long)k * 80 + f) * 512 + c;
      float w0 = wp[0], w1 = wp[512], w2 = wp[1024], w3 = wp[1536];
#pragma unroll
      for (int i = 0; i < 8; ++i) {
        v4f xv = *(const v4f*)&xs[(2 * i + k) * 80 + f];
        acc[i] += w0 * xv.x + w1 * xv.y + w2 * xv.z + w3 * xv.w;
      }
    }
  float inv = gamma[c] * rsqrtf(var[c] + BN_EPS);
  float sh  = beta[c] - mean[c] * inv;
  for (int i = 0; i < 8; ++i) {
    int t = t0 + i;
    if (t >= TOUT) break;
    float v = fmaxf(acc[i], 0.f) * inv + sh;
    long idx = ((long)b * TOUT + t) * CCH + c;
    outf[idx] = v;
    outb[idx] = (__bf16)v;
  }
}

// ---------------------------------------------------------------------------
// xp = Abf[ROWS x 512] @ Bpacked[512 x 1536] + bias.  WMMA bf16, f32 acc.
// WG 256 thr (8 waves, 2(M) x 4(N)), tile 128x128, double-buffered A staging.
// ---------------------------------------------------------------------------
__global__ __launch_bounds__(256) void xp_gemm(
    const __bf16* __restrict__ Abf, const __bf16* __restrict__ Bpk,
    const float* __restrict__ bias, float* __restrict__ Out) {
  __shared__ __bf16 Atile[2][128 * 32];
  const int tid = threadIdx.x, lane = tid & 31, w = tid >> 5;
  const int mwave = w >> 2, nwave = w & 3;
  const int blockM = blockIdx.y, blockN = blockIdx.x;
  const int lm = lane & 15, hs = lane >> 4;
  v8f acc[4][2] = {};

  auto stage = [&](int kt, int buf) {
#pragma unroll
    for (int it = 0; it < 2; ++it) {
      int r = (tid >> 2) + it * 64;
      int seg = (tid & 3);
      long grow = (long)blockM * 128 + r;
      v8bf val;
      if (grow < ROWS)
        val = *(const v8bf*)(Abf + grow * 512 + kt * 32 + seg * 8);
      else {
#pragma unroll
        for (int q = 0; q < 8; ++q) val[q] = (__bf16)0.0f;
      }
      *(v8bf*)(&Atile[buf][r * 32 + seg * 8]) = val;
    }
  };

  stage(0, 0);
  __syncthreads();
  for (int kt = 0; kt < KT16; ++kt) {
    const int cur = kt & 1;
    if (kt + 1 < KT16) stage(kt + 1, cur ^ 1);  // overlap with this tile's WMMAs
    v16bf afr[4];
#pragma unroll
    for (int i = 0; i < 4; ++i) {
      const __bf16* p =
          &Atile[cur][(mwave * 64 + i * 16 + lm) * 32 + (hs ? 8 : 0)];
      afr[i] = join8(*(const v8bf*)p, *(const v8bf*)(p + 16));
    }
#pragma unroll
    for (int j = 0; j < 2; ++j) {
      int nt = blockN * 8 + nwave * 2 + j;
      v16bf bfr = *(const v16bf*)(Bpk + (((long)kt * NT96 + nt) << 9) + (lane << 4));
#pragma unroll
      for (int i = 0; i < 4; ++i) acc[i][j] = wmma_bf16(afr[i], bfr, acc[i][j]);
    }
    __syncthreads();
  }
  // epilogue: +bias, store f32 (C layout: lane=N, vgpr j = M(+8 for hi lanes))
#pragma unroll
  for (int i = 0; i < 4; ++i) {
    long rowb = (long)blockM * 128 + mwave * 64 + i * 16;
#pragma unroll
    for (int j = 0; j < 2; ++j) {
      int colb = blockN * 128 + nwave * 32 + j * 16;
      float bv = bias[colb + lm];
#pragma unroll
      for (int jj = 0; jj < 8; ++jj) {
        long r = rowb + jj + (hs ? 8 : 0);
        if (r < ROWS) Out[r * N3 + colb + lm] = acc[i][j][jj] + bv;
      }
    }
  }
}

// ---------------------------------------------------------------------------
// GRU scan. 8 blocks x 256 thr: dir = bx>>2 (fwd/bwd), g = bx&3 (N-split).
// Wave w owns h columns [g*128+w*16, +16) in all 3 gates -> gate math runs
// directly on the WMMA accumulators. h state f32 in registers; bf16 h shared
// across the 4 WGs of a direction via ping-pong L2 buffers + atomic barrier.
// xp loads issued before the 96-WMMA GEMM; next step's xp prefetched.
// ---------------------------------------------------------------------------
__global__ __launch_bounds__(256) void gru_scan(
    const __bf16* __restrict__ prk_f, const __bf16* __restrict__ prk_b,
    const float* __restrict__ xp_f, const float* __restrict__ xp_b,
    const float* __restrict__ rb_f, const float* __restrict__ rb_b,
    float* __restrict__ ys_f, float* __restrict__ ys_b, __bf16* hglob,
    unsigned int* ctr) {
  __shared__ __bf16 hbuf[32 * 512];  // 32KB: h_{t-1} as bf16 [b][u]
  const int tid = threadIdx.x, lane = tid & 31, w = tid >> 5;
  const int bx = blockIdx.x, dir = bx >> 2, g = bx & 3;
  const __bf16* prk = dir ? prk_b : prk_f;
  const float* xp = dir ? xp_b : xp_f;
  const float* rb = dir ? rb_b : rb_f;
  float* ys = dir ? ys_b : ys_f;
  __bf16* hg = hglob + (long)dir * 2 * 16384;
  unsigned int* myc = ctr + dir * 64;
  const int lm = lane & 15, hs = lane >> 4;
  const int ucol = g * 128 + w * 16 + lm;
  const float rbz = rb[ucol], rbr = rb[512 + ucol], rbh = rb[1024 + ucol];

  for (int i = tid; i < 32 * 512 / 8; i += 256) {
    v8bf z;
#pragma unroll
    for (int q = 0; q < 8; ++q) z[q] = (__bf16)0.0f;
    ((v8bf*)hbuf)[i] = z;
  }
  __syncthreads();

  v8f hstate[2] = {};
  for (int s = 0; s < TOUT; ++s) {
    const int t = dir ? (TOUT - 1 - s) : s;
    // issue this step's xp loads early; their latency hides under the GEMM
    float px[3][2][8];
#pragma unroll
    for (int m = 0; m < 2; ++m)
#pragma unroll
      for (int jj = 0; jj < 8; ++jj) {
        int b = m * 16 + jj + (hs ? 8 : 0);
        long rowoff = ((long)b * TOUT + t) * N3 + ucol;
        px[0][m][jj] = xp[rowoff];
        px[1][m][jj] = xp[rowoff + 512];
        px[2][m][jj] = xp[rowoff + 1024];
      }
    v8f acc[3][2] = {};  // [gate z/r/h][m-tile]
#pragma unroll 4
    for (int kt = 0; kt < KT16; ++kt) {
      const __bf16* p0 = hbuf + lm * 512 + kt * 32 + (hs ? 8 : 0);
      v16bf a0 = join8(*(const v8bf*)p0, *(const v8bf*)(p0 + 16));
      const __bf16* p1 = p0 + 16 * 512;
      v16bf a1 = join8(*(const v8bf*)p1, *(const v8bf*)(p1 + 16));
#pragma unroll
      for (int e = 0; e < 3; ++e) {
        int nt = e * 32 + g * 8 + w;
        v16bf bfr =
            *(const v16bf*)(prk + (((long)kt * NT96 + nt) << 9) + (lane << 4));
        acc[e][0] = wmma_bf16(a0, bfr, acc[e][0]);
        acc[e][1] = wmma_bf16(a1, bfr, acc[e][1]);
      }
    }
    // gates + state update + output store (on accumulator layout)
    __bf16* hdst = hg + (s & 1) * 16384;
#pragma unroll
    for (int m = 0; m < 2; ++m) {
#pragma unroll
      for (int jj = 0; jj < 8; ++jj) {
        int b = m * 16 + jj + (hs ? 8 : 0);
        float z = sigmoidf(px[0][m][jj] + acc[0][m][jj] + rbz);
        float r = sigmoidf(px[1][m][jj] + acc[1][m][jj] + rbr);
        float hc = fmaxf(px[2][m][jj] + r * (acc[2][m][jj] + rbh), 0.f);
        float hn = z * hstate[m][jj] + (1.f - z) * hc;
        hstate[m][jj] = hn;
        ys[((long)b * TOUT + t) * UU + ucol] = hn;
        hdst[b * 512 + ucol] = (__bf16)hn;
      }
    }
    // cross-WG step barrier (release stores, arrive, spin, acquire)
    __threadfence();
    __syncthreads();
    if (tid == 0) {
      __hip_atomic_fetch_add(myc, 1u, __ATOMIC_RELEASE, __HIP_MEMORY_SCOPE_AGENT);
      unsigned int tgt = 4u * (unsigned)(s + 1);
      while (__hip_atomic_load(myc, __ATOMIC_ACQUIRE, __HIP_MEMORY_SCOPE_AGENT) < tgt)
        __builtin_amdgcn_s_sleep(1);
    }
    __syncthreads();
    __builtin_amdgcn_fence(__ATOMIC_ACQUIRE, "agent");
    // pull full h_t (all 4 WG slices) into LDS for next step's A fragments
    {
      const v4u* src = (const v4u*)hdst;
      v4u* dst = (v4u*)hbuf;
      for (int i = tid; i < 2048; i += 256) dst[i] = src[i];
    }
    // prefetch next step's xp rows (global_prefetch_b8; no counter impact)
    if (s + 1 < TOUT) {
      const int tn = dir ? (TOUT - 2 - s) : (s + 1);
#pragma unroll
      for (int m = 0; m < 2; ++m)
#pragma unroll
        for (int jj = 0; jj < 8; ++jj) {
          int b = m * 16 + jj + (hs ? 8 : 0);
          const float* p = xp + ((long)b * TOUT + tn) * N3 + ucol;
          __builtin_prefetch(p, 0, 3);
          __builtin_prefetch(p + 512, 0, 3);
          __builtin_prefetch(p + 1024, 0, 3);
        }
    }
    __syncthreads();
  }
}

// ---------------------------------------------------------------------------
__global__ void combine_dirs(const v4f* __restrict__ a, const v4f* __restrict__ b,
                             v4f* __restrict__ of, v4bf* __restrict__ ob, long n4) {
  long i = (long)blockIdx.x * 256 + threadIdx.x;
  if (i < n4) {
    v4f v = a[i] + b[i];
    of[i] = v;
    v4bf o;
#pragma unroll
    for (int q = 0; q < 4; ++q) o[q] = (__bf16)v[q];
    ob[i] = o;
  }
}

// ---------------------------------------------------------------------------
// BN + Dense(512->29) + softmax. One wave per (b,t) row; lane v = class.
// ---------------------------------------------------------------------------
__global__ __launch_bounds__(256) void bn_dense_softmax(
    const float* __restrict__ h, const float* __restrict__ gamma,
    const float* __restrict__ beta, const float* __restrict__ mean,
    const float* __restrict__ var, const float* __restrict__ wd,
    const float* __restrict__ bd, float* __restrict__ out) {
  __shared__ float rows[8][512];
  const int tid = threadIdx.x, lane = tid & 31, w = tid >> 5;
  const long row = (long)blockIdx.x * 8 + w;
  const bool valid = row < ROWS;
  if (valid) {
    for (int i = lane; i < 512; i += 32) {
      float inv = gamma[i] * rsqrtf(var[i] + BN_EPS);
      rows[w][i] = h[row * 512 + i] * inv + (beta[i] - mean[i] * inv);
    }
  }
  __syncthreads();
  int v = (lane < NV) ? lane : 0;
  float acc = bd[v];
  for (int u = 0; u < 512; ++u) acc += rows[w][u] * wd[u * NV + v];
  float lg = (lane < NV) ? acc : -1e30f;
  float mx = lg;
#pragma unroll
  for (int off = 16; off; off >>= 1) mx = fmaxf(mx, __shfl_xor(mx, off, 32));
  float e = (lane < NV) ? __expf(lg - mx) : 0.f;
  float sm = e;
#pragma unroll
  for (int off = 16; off; off >>= 1) sm += __shfl_xor(sm, off, 32);
  if (valid && lane < NV) out[row * NV + lane] = e / sm;
}

// ---------------------------------------------------------------------------
extern "C" void kernel_launch(void* const* d_in, const int* in_sizes, int n_in,
                              void* d_out, int out_size, void* d_ws, size_t ws_size,
                              hipStream_t stream) {
  (void)in_sizes; (void)n_in; (void)out_size; (void)ws_size;
  const float* x      = (const float*)d_in[0];
  const float* conv_w = (const float*)d_in[1];
  const float* conv_b = (const float*)d_in[2];
  const float* k_f    = (const float*)d_in[3];
  const float* rk_f   = (const float*)d_in[4];
  const float* b_f    = (const float*)d_in[5];
  const float* k_b    = (const float*)d_in[6];
  const float* rk_b   = (const float*)d_in[7];
  const float* b_b    = (const float*)d_in[8];
  const float* gamma  = (const float*)d_in[9];
  const float* beta   = (const float*)d_in[10];
  const float* mean   = (const float*)d_in[11];
  const float* var    = (const float*)d_in[12];
  const float* dense_w = (const float*)d_in[13];
  const float* dense_b = (const float*)d_in[14];

  char* ws = (char*)d_ws;
  size_t off = 0;
  auto alloc = [&](size_t bytes) {
    void* p = ws + off;
    off += (bytes + 255) & ~(size_t)255;
    return p;
  };
  float*  cur  = (float*)alloc((size_t)ROWS * 512 * 4);
  __bf16* curb = (__bf16*)alloc((size_t)ROWS * 512 * 2);
  float*  xpf  = (float*)alloc((size_t)ROWS * N3 * 4);
  float*  xpb  = (float*)alloc((size_t)ROWS * N3 * 4);
  float*  ysf  = (float*)alloc((size_t)ROWS * 512 * 4);
  float*  ysb  = (float*)alloc((size_t)ROWS * 512 * 4);
  __bf16* pkf  = (__bf16*)alloc((size_t)512 * N3 * 2);
  __bf16* pkb  = (__bf16*)alloc((size_t)512 * N3 * 2);
  __bf16* prkf = (__bf16*)alloc((size_t)512 * N3 * 2);
  __bf16* prkb = (__bf16*)alloc((size_t)512 * N3 * 2);
  __bf16* hg   = (__bf16*)alloc((size_t)4 * 16384 * 2);  // 2 dirs x ping-pong
  unsigned int* ctr = (unsigned int*)alloc(512);

  pack_weights<<<6144, 256, 0, stream>>>(k_f, k_b, rk_f, rk_b, pkf, pkb, prkf, prkb);
  conv_bn<<<dim3(125, 32), 512, 0, stream>>>(x, conv_w, conv_b, gamma, beta, mean,
                                             var, cur, curb);
  for (int layer = 0; layer < 2; ++layer) {
    xp_gemm<<<dim3(12, 249), 256, 0, stream>>>(curb, pkf, b_f, xpf);
    xp_gemm<<<dim3(12, 249), 256, 0, stream>>>(curb, pkb, b_b, xpb);
    (void)hipMemsetAsync(ctr, 0, 512, stream);
    gru_scan<<<8, 256, 0, stream>>>(prkf, prkb, xpf, xpb, b_f + 1536, b_b + 1536,
                                    ysf, ysb, hg, ctr);
    combine_dirs<<<(int)(((long)ROWS * 512 / 4 + 255) / 256), 256, 0, stream>>>(
        (const v4f*)ysf, (const v4f*)ysb, (v4f*)cur, (v4bf*)curb,
        (long)ROWS * 512 / 4);
  }
  bn_dense_softmax<<<(ROWS + 7) / 8, 256, 0, stream>>>(
      cur, gamma, beta, mean, var, dense_w, dense_b, (float*)d_out);
}
